// Dim1_StructuredSparse_23381801960227
// MI455X (gfx1250) — matrix operand order repaired and checked by reading, not compiled, not run on hardware
//
#include <hip/hip_runtime.h>

#define N_OUT    4096
#define N_IN     4096
#define N_SPARSE 409
#define M_TOTAL  16384   // B * S = 4 * 4096

typedef __attribute__((ext_vector_type(16))) __bf16         v16bf;
typedef __attribute__((ext_vector_type(8)))  float          v8f;
typedef __attribute__((ext_vector_type(8)))  unsigned short v8u;
typedef __attribute__((ext_vector_type(16))) unsigned short v16u;

__device__ __forceinline__ unsigned short f2bf(float f) {
  // round-to-nearest-even f32 -> bf16
  unsigned int u = __builtin_bit_cast(unsigned int, f);
  u += 0x7FFFu + ((u >> 16) & 1u);
  return (unsigned short)(u >> 16);
}

// ---------------------------------------------------------------------------
// Kernel 1: zero-fill the 256 MB output (poisoned by harness) with b128 stores
// ---------------------------------------------------------------------------
__global__ __launch_bounds__(256) void zero_out_kernel(float4* __restrict__ out, int n4) {
  int i = blockIdx.x * blockDim.x + threadIdx.x;
  if (i < n4) out[i] = make_float4(0.f, 0.f, 0.f, 0.f);
}

// ---------------------------------------------------------------------------
// Kernel 2: BF16-WMMA GEMM (X @ W^T) with column scatter.
// Workgroup = 256 threads = 8 waves -> 64(M) x 32(N) tile, K staged 64 at a
// time through LDS (f32 -> bf16 on the fly). Each wave owns one 16x16 tile.
// ---------------------------------------------------------------------------
#define LDA 72   // padded row stride (ushorts) for 64-deep K panel
#define LDB 72
#define KSTEP 64

__global__ __launch_bounds__(256)
void sparse_proj_wmma(const float* __restrict__ x,
                      const float* __restrict__ w,
                      const int*   __restrict__ sidx,
                      float*       __restrict__ out) {
  __shared__ __align__(16) unsigned short As[64 * LDA];  // 64 rows of X, 64 K
  __shared__ __align__(16) unsigned short Bs[32 * LDB];  // 32 rows of W, 64 K

  const int tid  = threadIdx.x;
  const int lane = tid & 31;
  const int wave = tid >> 5;
  const int wm   = wave >> 1;             // 0..3 : M subtile
  const int wn   = wave & 1;              // 0..1 : N subtile
  const int m0   = blockIdx.y * 64;
  const int n0   = blockIdx.x * 32;

  v8f c = {};                              // f32 accumulator tile

  // Per-lane fragment base pointers (CDNA5 16-bit WMMA VGPR layouts):
  // A 16x32: lane<16 holds K lo-half of window, lane>=16 the hi-half (offset 8)
  // B 32x16: same K split as A (lane>=16 offset 8, hi 8 at +16)
  const int row16 = lane & 15;
  const unsigned short* aptr = &As[(wm * 16 + row16) * LDA + ((lane >> 4) << 3)];
  const unsigned short* bptr = &Bs[(wn * 16 + row16) * LDB + ((lane >> 4) << 3)];

  for (int k0 = 0; k0 < N_IN; k0 += KSTEP) {
    // ---- stage A panel: 64 rows x 64 K of x, f32 -> bf16 (coalesced b128) ----
#pragma unroll
    for (int i = 0; i < 4; ++i) {
      int idx = tid + i * 256;            // 0..1023 = 64 rows * 16 float4
      int r   = idx >> 4;
      int c4  = idx & 15;
      float4 v = *(const float4*)(&x[(size_t)(m0 + r) * N_IN + k0 + c4 * 4]);
      unsigned int p0 = (unsigned int)f2bf(v.x) | ((unsigned int)f2bf(v.y) << 16);
      unsigned int p1 = (unsigned int)f2bf(v.z) | ((unsigned int)f2bf(v.w) << 16);
      unsigned int* dst = (unsigned int*)&As[r * LDA + c4 * 4];
      dst[0] = p0; dst[1] = p1;
    }
    // ---- stage B panel: 32 W-rows x 64 K (zero-pad past row 409) ----
#pragma unroll
    for (int i = 0; i < 2; ++i) {
      int idx = tid + i * 256;            // 0..511 = 32 rows * 16 float4
      int r   = idx >> 4;
      int c4  = idx & 15;
      int wr  = n0 + r;
      float4 v = make_float4(0.f, 0.f, 0.f, 0.f);
      if (wr < N_SPARSE)
        v = *(const float4*)(&w[(size_t)wr * N_IN + k0 + c4 * 4]);
      unsigned int p0 = (unsigned int)f2bf(v.x) | ((unsigned int)f2bf(v.y) << 16);
      unsigned int p1 = (unsigned int)f2bf(v.z) | ((unsigned int)f2bf(v.w) << 16);
      unsigned int* dst = (unsigned int*)&Bs[r * LDB + c4 * 4];
      dst[0] = p0; dst[1] = p1;
    }
    __syncthreads();

#pragma unroll
    for (int kk = 0; kk < KSTEP; kk += 32) {
      v8u alo = *(const v8u*)(aptr + kk);        // K lo 8 of window
      v8u ahi = *(const v8u*)(aptr + kk + 16);   // K hi 8 of window
      v8u blo = *(const v8u*)(bptr + kk);
      v8u bhi = *(const v8u*)(bptr + kk + 16);
      v16u au = __builtin_shufflevector(alo, ahi, 0,1,2,3,4,5,6,7,8,9,10,11,12,13,14,15);
      v16u bu = __builtin_shufflevector(blo, bhi, 0,1,2,3,4,5,6,7,8,9,10,11,12,13,14,15);
      v16bf a = __builtin_bit_cast(v16bf, au);
      v16bf b = __builtin_bit_cast(v16bf, bu);
      c = __builtin_amdgcn_wmma_f32_16x16x32_bf16(false, a, false, b,
                                                  (short)0, c, false, false);
    }
    __syncthreads();
  }

  // ---- epilogue: scatter 16x16 f32 tile into sparse_idx columns ----
  // C layout: VGPR r -> M = r (lanes 0-15) / r+8 (lanes 16-31); N = lane%16
  const int nloc = n0 + wn * 16 + row16;
  if (nloc < N_SPARSE) {
    const int col   = sidx[nloc];
    const int rbase = m0 + wm * 16 + ((lane >> 4) << 3);
#pragma unroll
    for (int r = 0; r < 8; ++r) {
      out[(size_t)(rbase + r) * N_OUT + col] = c[r];
    }
  }
}

// ---------------------------------------------------------------------------
extern "C" void kernel_launch(void* const* d_in, const int* in_sizes, int n_in,
                              void* d_out, int out_size, void* d_ws, size_t ws_size,
                              hipStream_t stream) {
  const float* x    = (const float*)d_in[0];   // [4, 4096, 4096] f32
  // d_in[1] = y (zeros) -- unused; we zero-fill d_out ourselves
  const float* w    = (const float*)d_in[2];   // [409, 4096] f32
  const int*   sidx = (const int*)d_in[3];     // [409] i32
  float*       out  = (float*)d_out;           // [4, 4096, 4096] f32

  // 1) zero-fill output (harness poisons it)
  int n4 = out_size >> 2;
  zero_out_kernel<<<dim3((n4 + 255) / 256), dim3(256), 0, stream>>>((float4*)out, n4);

  // 2) WMMA GEMM + scatter. Grid ordered N-fastest so all N-blocks sharing an
  //    X band are adjacent -> X re-reads served from the 192 MB L2.
  dim3 grid((N_SPARSE + 31) / 32, M_TOTAL / 64);   // (13, 256)
  sparse_proj_wmma<<<grid, dim3(256), 0, stream>>>(x, w, sidx, out);
}
